// RoutedBitLinear_43928925503764
// MI455X (gfx1250) — matrix operand order repaired and checked by reading, not compiled, not run on hardware
//
#include <hip/hip_runtime.h>
#include <hip/hip_bf16.h>
#include <stdint.h>

typedef __attribute__((ext_vector_type(16))) _Float16 v16h;
typedef __attribute__((ext_vector_type(8)))  _Float16 v8h;
typedef __attribute__((ext_vector_type(4)))  _Float16 v4h;
typedef __attribute__((ext_vector_type(8)))  float    v8f;

// Problem constants (fixed by setup_inputs in the reference)
#define M_DIM 8192     // B*S = 4*2048
#define N_DIM 4096     // out_features
#define K_DIM 4096     // in_features
#define GS    128      // group_size
#define GPR   (K_DIM / GS)   // 32 scale groups per output row

// Block tile: 256x128, 8 wave32s, each wave owns 64x64 (4x4 WMMA accums)
#define BM 256
#define BN 128

// Fast (async) path: K-step 64, f16 sources from workspace
#define BKF 64
#define NKF (K_DIM / BKF)    // 64 K-steps
#define LDF 72               // halfs per LDS row (64 + 8 pad, 144B stride)

// Fallback path: K-step 32, fp32 sources converted in-kernel
#define BK  32
#define NK  (K_DIM / BK)
#define LDA 40
#define LDB 40

// ---------------------------------------------------------------------------
__device__ __forceinline__ uint32_t lds_off(const void* p) {
    // Generic pointers to LDS carry the byte offset in the low 32 bits.
    return (uint32_t)(uintptr_t)p;
}

// Issue one 16B-per-lane async DMA copy global->LDS (GVS mode, ASYNCcnt).
#define ASYNC_CP16(ldsoff, goff32, sbase)                                     \
    asm volatile("global_load_async_to_lds_b128 %0, %1, %2"                   \
                 :: "v"(ldsoff), "v"(goff32), "s"(sbase) : "memory")

// ---------------------------------------------------------------------------
// Kernel 1: blended[g] = sum_p routing[p] * profile_scales[p, g]
// ---------------------------------------------------------------------------
__global__ void blend_scales_kernel(const float* __restrict__ profile_scales,
                                    const float* __restrict__ routing,
                                    float* __restrict__ blended,
                                    int G, int P) {
    int g = blockIdx.x * blockDim.x + threadIdx.x;
    if (g >= G) return;
    float acc = 0.0f;
    #pragma unroll 8
    for (int p = 0; p < P; ++p)
        acc += routing[p] * profile_scales[(size_t)p * G + g];
    blended[g] = acc;
}

// ---------------------------------------------------------------------------
// Kernel 2a: xh = f16(x)   (8 floats / thread, 16B stores)
// ---------------------------------------------------------------------------
__global__ __launch_bounds__(256)
void convert_x_f16_kernel(const float* __restrict__ x, _Float16* __restrict__ xh) {
    size_t i = ((size_t)blockIdx.x * 256 + threadIdx.x) * 8;
    float4 a = *(const float4*)(x + i);
    float4 b = *(const float4*)(x + i + 4);
    v8h o = { (_Float16)a.x, (_Float16)a.y, (_Float16)a.z, (_Float16)a.w,
              (_Float16)b.x, (_Float16)b.y, (_Float16)b.z, (_Float16)b.w };
    *(v8h*)(xh + i) = o;
}

// ---------------------------------------------------------------------------
// Kernel 2b: wh[n,k] = f16( signs[n,k] * blended[n*GPR + k/GS] )
// ---------------------------------------------------------------------------
__global__ __launch_bounds__(256)
void build_w_f16_kernel(const float* __restrict__ signs,
                        const float* __restrict__ blended,
                        _Float16* __restrict__ wh) {
    size_t i = ((size_t)blockIdx.x * 256 + threadIdx.x) * 8;   // 8 k-elems, one group
    int n = (int)(i >> 12);          // / K_DIM
    int k = (int)(i & (K_DIM - 1));
    float s = blended[(size_t)n * GPR + (k >> 7)];
    float4 a = *(const float4*)(signs + i);
    float4 b = *(const float4*)(signs + i + 4);
    v8h o = { (_Float16)(a.x * s), (_Float16)(a.y * s),
              (_Float16)(a.z * s), (_Float16)(a.w * s),
              (_Float16)(b.x * s), (_Float16)(b.y * s),
              (_Float16)(b.z * s), (_Float16)(b.w * s) };
    *(v8h*)(wh + i) = o;
}

// ---------------------------------------------------------------------------
// Fragment loads (wave32 WMMA 16-bit layouts, ISA 7.12.2); `ld` = row stride.
// ---------------------------------------------------------------------------
__device__ __forceinline__ v16h load_a_frag(const _Float16* __restrict__ tile_row0,
                                            int ld, int hl, int ml) {
    const _Float16* row = tile_row0 + ml * ld;
    v8h lo = *(const v8h*)(row + hl * 8);
    v8h hi = *(const v8h*)(row + 16 + hl * 8);
    v16h a;
    #pragma unroll
    for (int i = 0; i < 8; ++i) { a[i] = lo[i]; a[i + 8] = hi[i]; }
    return a;
}

__device__ __forceinline__ v16h load_b_frag(const _Float16* __restrict__ tile_row0,
                                            int ld, int hl, int ml) {
    const _Float16* row = tile_row0 + ml * ld;
    v8h lo = *(const v8h*)(row + hl * 8);
    v8h hi = *(const v8h*)(row + 16 + hl * 8);
    v16h b;
    #pragma unroll
    for (int i = 0; i < 8; ++i) { b[i] = lo[i]; b[i + 8] = hi[i]; }
    return b;
}

// ---------------------------------------------------------------------------
// Kernel 3 (fast): out = xh * wh^T, async-DMA staged, BK=64, 32 WMMAs/K-step.
// ---------------------------------------------------------------------------
__global__ __launch_bounds__(256)
void routed_bitlinear_wmma_async_kernel(const _Float16* __restrict__ xh,
                                        const _Float16* __restrict__ wh,
                                        float* __restrict__ out) {
    __shared__ _Float16 As[2][BM * LDF];   // 2 x 36864 B
    __shared__ _Float16 Bs[2][BN * LDF];   // 2 x 18432 B

    const int tid  = threadIdx.x;
    const int m0   = blockIdx.y * BM;
    const int n0   = blockIdx.x * BN;
    const int wave = tid >> 5, lane = tid & 31;
    const int wm   = wave >> 1, wn = wave & 1;
    const int hl   = lane >> 4, ml = lane & 15;

    // DMA chunk assignment: 16B per lane per issue; per-thread fixed
    // (row0 = tid>>3 stepping by 32 rows per issue, column chunk kc0 = tid&7).
    const int r0  = tid >> 3;
    const int kc0 = tid & 7;

    const uint64_t xbase = (uint64_t)(uintptr_t)xh;
    const uint64_t wbase = (uint64_t)(uintptr_t)wh;

    uint32_t ldsA[2], ldsB[2];
    ldsA[0] = lds_off(&As[0][0]) + r0 * (LDF * 2) + kc0 * 16;
    ldsA[1] = lds_off(&As[1][0]) + r0 * (LDF * 2) + kc0 * 16;
    ldsB[0] = lds_off(&Bs[0][0]) + r0 * (LDF * 2) + kc0 * 16;
    ldsB[1] = lds_off(&Bs[1][0]) + r0 * (LDF * 2) + kc0 * 16;

    const uint32_t goA0 = (uint32_t)(((uint32_t)(m0 + r0) * K_DIM + kc0 * 8) * 2);
    const uint32_t goB0 = (uint32_t)(((uint32_t)(n0 + r0) * K_DIM + kc0 * 8) * 2);
    const uint32_t rowStep = 32u * K_DIM * 2;   // 32 rows per issue index

    v8f acc[4][4];
    #pragma unroll
    for (int tm = 0; tm < 4; ++tm)
        #pragma unroll
        for (int tn = 0; tn < 4; ++tn)
            acc[tm][tn] = (v8f){0.f, 0.f, 0.f, 0.f, 0.f, 0.f, 0.f, 0.f};

    // ---- prologue: DMA K-tile 0 into buffer 0 ----
    #pragma unroll
    for (int i = 0; i < 8; ++i)
        ASYNC_CP16(ldsA[0] + i * (32 * LDF * 2), goA0 + i * rowStep, xbase);
    #pragma unroll
    for (int i = 0; i < 4; ++i)
        ASYNC_CP16(ldsB[0] + i * (32 * LDF * 2), goB0 + i * rowStep, wbase);
    asm volatile("s_wait_asynccnt 0x0" ::: "memory");
    __syncthreads();

    for (int kt = 0; kt < NKF; ++kt) {
        const int cur = kt & 1;
        const int nxt = cur ^ 1;

        // ---- DMA K-tile kt+1 into `nxt` (overlaps with WMMAs below) ----
        if (kt + 1 < NKF) {
            const uint32_t kb = (uint32_t)(kt + 1) * (BKF * 2);   // byte advance along K
            #pragma unroll
            for (int i = 0; i < 8; ++i)
                ASYNC_CP16(ldsA[nxt] + i * (32 * LDF * 2), goA0 + kb + i * rowStep, xbase);
            #pragma unroll
            for (int i = 0; i < 4; ++i)
                ASYNC_CP16(ldsB[nxt] + i * (32 * LDF * 2), goB0 + kb + i * rowStep, wbase);
        }

        // ---- compute on `cur`: 2 K-substeps x 16 = 32 WMMAs per wave ----
        #pragma unroll
        for (int ks = 0; ks < 2; ++ks) {
            v16h bf[4];
            #pragma unroll
            for (int tn = 0; tn < 4; ++tn)
                bf[tn] = load_b_frag(&Bs[cur][(wn * 64 + tn * 16) * LDF + ks * 32],
                                     LDF, hl, ml);
            #pragma unroll
            for (int tm = 0; tm < 4; ++tm) {
                v16h a = load_a_frag(&As[cur][(wm * 64 + tm * 16) * LDF + ks * 32],
                                     LDF, hl, ml);
                #pragma unroll
                for (int tn = 0; tn < 4; ++tn)
                    acc[tm][tn] = __builtin_amdgcn_wmma_f32_16x16x32_f16(
                        false, a, false, bf[tn], (short)0, acc[tm][tn], false, false);
            }
        }

        asm volatile("s_wait_asynccnt 0x0" ::: "memory");
        __syncthreads();   // one barrier per K-step (ping-pong)
    }

    // ---- epilogue: lane l holds (M = hl*8 + r, N = ml) per 16x16 tile ----
    #pragma unroll
    for (int tm = 0; tm < 4; ++tm) {
        #pragma unroll
        for (int tn = 0; tn < 4; ++tn) {
            const int rowbase = m0 + wm * 64 + tm * 16 + hl * 8;
            const int col     = n0 + wn * 64 + tn * 16 + ml;
            #pragma unroll
            for (int r = 0; r < 8; ++r)
                out[(size_t)(rowbase + r) * N_DIM + col] = acc[tm][tn][r];
        }
    }
}

// ---------------------------------------------------------------------------
// Kernel 3 (fallback): fp32 sources, in-kernel f16 conversion, BK=32.
// ---------------------------------------------------------------------------
__global__ __launch_bounds__(256)
void routed_bitlinear_wmma_fallback(const float* __restrict__ x,
                                    const float* __restrict__ signs,
                                    const float* __restrict__ bscale,
                                    float* __restrict__ out) {
    __shared__ _Float16 As[2][BM * LDA];
    __shared__ _Float16 Bs[2][BN * LDB];

    const int tid  = threadIdx.x;
    const int m0   = blockIdx.y * BM;
    const int n0   = blockIdx.x * BN;
    const int wave = tid >> 5, lane = tid & 31;
    const int wm   = wave >> 1, wn = wave & 1;
    const int hl   = lane >> 4, ml = lane & 15;

    v8f acc[4][4];
    #pragma unroll
    for (int tm = 0; tm < 4; ++tm)
        #pragma unroll
        for (int tn = 0; tn < 4; ++tn)
            acc[tm][tn] = (v8f){0.f, 0.f, 0.f, 0.f, 0.f, 0.f, 0.f, 0.f};

    float4 a_reg[8], b_reg[4];
    float  s_reg[4];

    #pragma unroll
    for (int i = 0; i < 8; ++i) {
        int c = i * 256 + tid, row = c >> 3, kc = c & 7;
        a_reg[i] = *(const float4*)(x + (size_t)(m0 + row) * K_DIM + kc * 4);
    }
    #pragma unroll
    for (int i = 0; i < 4; ++i) {
        int c = i * 256 + tid, row = c >> 3, kc = c & 7;
        b_reg[i] = *(const float4*)(signs + (size_t)(n0 + row) * K_DIM + kc * 4);
        s_reg[i] = bscale[(size_t)(n0 + row) * GPR];
    }
    #pragma unroll
    for (int i = 0; i < 8; ++i) {
        int c = i * 256 + tid, row = c >> 3, kc = c & 7;
        v4h va = { (_Float16)a_reg[i].x, (_Float16)a_reg[i].y,
                   (_Float16)a_reg[i].z, (_Float16)a_reg[i].w };
        *(v4h*)(&As[0][row * LDA + kc * 4]) = va;
    }
    #pragma unroll
    for (int i = 0; i < 4; ++i) {
        int c = i * 256 + tid, row = c >> 3, kc = c & 7;
        const float s = s_reg[i];
        v4h vb = { (_Float16)(b_reg[i].x * s), (_Float16)(b_reg[i].y * s),
                   (_Float16)(b_reg[i].z * s), (_Float16)(b_reg[i].w * s) };
        *(v4h*)(&Bs[0][row * LDB + kc * 4]) = vb;
    }
    __syncthreads();

    for (int kt = 0; kt < NK; ++kt) {
        const int  cur      = kt & 1;
        const int  nxt      = cur ^ 1;
        const bool has_next = (kt + 1) < NK;
        const int  k1       = (kt + 1) * BK;

        if (has_next) {
            const int grp = k1 >> 7;
            #pragma unroll
            for (int i = 0; i < 8; ++i) {
                int c = i * 256 + tid, row = c >> 3, kc = c & 7;
                a_reg[i] = *(const float4*)(x + (size_t)(m0 + row) * K_DIM + k1 + kc * 4);
            }
            #pragma unroll
            for (int i = 0; i < 4; ++i) {
                int c = i * 256 + tid, row = c >> 3, kc = c & 7;
                b_reg[i] = *(const float4*)(signs + (size_t)(n0 + row) * K_DIM + k1 + kc * 4);
                s_reg[i] = bscale[(size_t)(n0 + row) * GPR + grp];
            }
        }

        {
            v16h bf[4];
            #pragma unroll
            for (int tn = 0; tn < 4; ++tn)
                bf[tn] = load_b_frag(&Bs[cur][(wn * 64 + tn * 16) * LDB], LDB, hl, ml);
            #pragma unroll
            for (int tm = 0; tm < 4; ++tm) {
                v16h a = load_a_frag(&As[cur][(wm * 64 + tm * 16) * LDA], LDA, hl, ml);
                #pragma unroll
                for (int tn = 0; tn < 4; ++tn)
                    acc[tm][tn] = __builtin_amdgcn_wmma_f32_16x16x32_f16(
                        false, a, false, bf[tn], (short)0, acc[tm][tn], false, false);
            }
        }

        if (has_next) {
            #pragma unroll
            for (int i = 0; i < 8; ++i) {
                int c = i * 256 + tid, row = c >> 3, kc = c & 7;
                v4h va = { (_Float16)a_reg[i].x, (_Float16)a_reg[i].y,
                           (_Float16)a_reg[i].z, (_Float16)a_reg[i].w };
                *(v4h*)(&As[nxt][row * LDA + kc * 4]) = va;
            }
            #pragma unroll
            for (int i = 0; i < 4; ++i) {
                int c = i * 256 + tid, row = c >> 3, kc = c & 7;
                const float s = s_reg[i];
                v4h vb = { (_Float16)(b_reg[i].x * s), (_Float16)(b_reg[i].y * s),
                           (_Float16)(b_reg[i].z * s), (_Float16)(b_reg[i].w * s) };
                *(v4h*)(&Bs[nxt][row * LDB + kc * 4]) = vb;
            }
        }
        __syncthreads();
    }

    #pragma unroll
    for (int tm = 0; tm < 4; ++tm) {
        #pragma unroll
        for (int tn = 0; tn < 4; ++tn) {
            const int rowbase = m0 + wm * 64 + tm * 16 + hl * 8;
            const int col     = n0 + wn * 64 + tn * 16 + ml;
            #pragma unroll
            for (int r = 0; r < 8; ++r)
                out[(size_t)(rowbase + r) * N_DIM + col] = acc[tm][tn][r];
        }
    }
}

// ---------------------------------------------------------------------------
extern "C" void kernel_launch(void* const* d_in, const int* in_sizes, int n_in,
                              void* d_out, int out_size, void* d_ws, size_t ws_size,
                              hipStream_t stream) {
    const float* x       = (const float*)d_in[0];  // [8192, 4096]
    const float* signs   = (const float*)d_in[1];  // [4096, 4096]
    const float* pscales = (const float*)d_in[2];  // [P, G]
    const float* routing = (const float*)d_in[3];  // [P]

    const int P = in_sizes[3];
    const int G = in_sizes[2] / P;                 // 131072

    // Workspace layout: [blended G*4B][pad to 1MB][xh 64MB][wh 32MB]
    const size_t XH_OFF = 1u << 20;
    const size_t XH_SZ  = (size_t)M_DIM * K_DIM * sizeof(_Float16);  // 64 MB
    const size_t WH_SZ  = (size_t)N_DIM * K_DIM * sizeof(_Float16);  // 32 MB
    const size_t NEED   = XH_OFF + XH_SZ + WH_SZ;

    float*     blended = (float*)d_ws;
    _Float16*  xh      = (_Float16*)((char*)d_ws + XH_OFF);
    _Float16*  wh      = (_Float16*)((char*)d_ws + XH_OFF + XH_SZ);

    blend_scales_kernel<<<(G + 255) / 256, 256, 0, stream>>>(
        pscales, routing, blended, G, P);

    dim3 grid(N_DIM / BN, M_DIM / BM);             // (32, 32)

    if (ws_size >= NEED) {
        convert_x_f16_kernel<<<(M_DIM * (K_DIM / 8)) / 256, 256, 0, stream>>>(x, xh);
        build_w_f16_kernel<<<(N_DIM * (K_DIM / 8)) / 256, 256, 0, stream>>>(signs, blended, wh);
        routed_bitlinear_wmma_async_kernel<<<grid, 256, 0, stream>>>(
            xh, wh, (float*)d_out);
    } else {
        routed_bitlinear_wmma_fallback<<<grid, 256, 0, stream>>>(
            x, signs, blended, (float*)d_out);
    }
}
